// MoeLayer_32461362823513
// MI455X (gfx1250) — matrix operand order repaired and checked by reading, not compiled, not run on hardware
//
#include <hip/hip_runtime.h>
#include <hip/hip_bf16.h>
#include <stdint.h>

#define LDIM 1024
#define BDIM 4
#define DDIM 1024
#define EEXP 8
#define KTOP 2
#define HDIM 4096
#define NTOK (LDIM * BDIM)        // 4096 tokens
#define NENT (NTOK * KTOP)        // 8192 (token, k) entries
#define PADR 32                   // bucket row padding (= wave M-tile)
#define SLOT_MAX 8448             // 8192 + 8*31 headroom, rounded: 264*32
#define NTILES (SLOT_MAX / PADR)  // 264 row tiles

typedef __attribute__((ext_vector_type(16))) __bf16 v16bf;
typedef __attribute__((ext_vector_type(8)))  __bf16 v8bf;
typedef __attribute__((ext_vector_type(8)))  float  v8f;

// ---------------- helpers ----------------

__device__ __forceinline__ unsigned short f2bf(float f) {
    unsigned u = __builtin_bit_cast(unsigned, f);
    if ((u & 0x7fffffffu) > 0x7f800000u)      // NaN: keep quiet-NaN bits
        return (unsigned short)((u >> 16) | 0x40);
    u += 0x7fffu + ((u >> 16) & 1u);          // round-to-nearest-even
    return (unsigned short)(u >> 16);
}

__device__ __forceinline__ float gelu_tanh(float x) {
    float x3 = x * x * x;
    float t = tanhf(0.7978845608028654f * (x + 0.044715f * x3));
    return 0.5f * x * (1.0f + t);
}

// A fragment (16x32 bf16, row-major source): lane holds row lane&15;
// K-half ko = (lane>>4)*8: elems [0..7] = K[ko..ko+7], [8..15] = K[16+ko..16+ko+7]
__device__ __forceinline__ v16bf load_afrag(const unsigned short* p) {
    v8bf lo = *reinterpret_cast<const v8bf*>(p);
    v8bf hi = *reinterpret_cast<const v8bf*>(p + 16);
    return __builtin_shufflevector(lo, hi, 0, 1, 2, 3, 4, 5, 6, 7,
                                   8, 9, 10, 11, 12, 13, 14, 15);
}

#define WMMA_BF16(A, B, C) \
    __builtin_amdgcn_wmma_f32_16x16x32_bf16(false, (A), false, (B), (short)0, (C), false, false)

// ---------------- small kernels ----------------

__global__ void k_zero_counts(int* counts) {
    if (threadIdx.x < EEXP) counts[threadIdx.x] = 0;
}

__global__ void k_cvt_bf16(const float* __restrict__ src,
                           unsigned short* __restrict__ dst, int n4) {
    int i = blockIdx.x * blockDim.x + threadIdx.x;
    int stride = gridDim.x * blockDim.x;
    for (; i < n4; i += stride) {
        float4 v = reinterpret_cast<const float4*>(src)[i];
        ushort4 o;
        o.x = f2bf(v.x); o.y = f2bf(v.y); o.z = f2bf(v.z); o.w = f2bf(v.w);
        reinterpret_cast<ushort4*>(dst)[i] = o;
    }
}

// task_logits[b][e] = task_param[b] @ task_gate_w[:,e] + task_gate_b[e]
__global__ __launch_bounds__(256)
void k_task_logits(const float* __restrict__ tp, const float* __restrict__ tw,
                   const float* __restrict__ tb, float* __restrict__ tlog) {
    __shared__ float red[256 * EEXP];
    const int b = blockIdx.x, tid = threadIdx.x;
    float acc[EEXP] = {};
    float4 xv = reinterpret_cast<const float4*>(tp + (size_t)b * DDIM)[tid];
    const float* gr = tw + (size_t)tid * 4 * EEXP;
    float xs[4] = {xv.x, xv.y, xv.z, xv.w};
#pragma unroll
    for (int j = 0; j < 4; j++)
#pragma unroll
        for (int e = 0; e < EEXP; e++) acc[e] += xs[j] * gr[j * EEXP + e];
#pragma unroll
    for (int e = 0; e < EEXP; e++) red[tid * EEXP + e] = acc[e];
    __syncthreads();
    for (int s = 128; s > 0; s >>= 1) {
        if (tid < s)
#pragma unroll
            for (int e = 0; e < EEXP; e++)
                red[tid * EEXP + e] += red[(tid + s) * EEXP + e];
        __syncthreads();
    }
    if (tid == 0)
#pragma unroll
        for (int e = 0; e < EEXP; e++) tlog[b * EEXP + e] = red[e] + tb[e];
}

// Per-token gating: logits, top-2, renormalized weights, expert counts.
__global__ __launch_bounds__(256)
void k_gate(const float* __restrict__ x,
            const float* __restrict__ gw_img, const float* __restrict__ gb_img,
            const float* __restrict__ gw_txt, const float* __restrict__ gb_txt,
            const float* __restrict__ alphap, const int* __restrict__ is_textp,
            const float* __restrict__ tlog, float* __restrict__ glog,
            int* __restrict__ sel, float* __restrict__ wgt, int* __restrict__ counts) {
    __shared__ float red[256 * EEXP];
    const int t = blockIdx.x, tid = threadIdx.x;
    const int b = t & (BDIM - 1);
    const int txt = is_textp[0];
    const float* gw = txt ? gw_txt : gw_img;
    const float* gb = txt ? gb_txt : gb_img;
    const float alpha = alphap[0];

    float acc[EEXP] = {};
    float4 xv = reinterpret_cast<const float4*>(x + (size_t)t * DDIM)[tid];
    const float* gr = gw + (size_t)tid * 4 * EEXP;
    float xs[4] = {xv.x, xv.y, xv.z, xv.w};
#pragma unroll
    for (int j = 0; j < 4; j++)
#pragma unroll
        for (int e = 0; e < EEXP; e++) acc[e] += xs[j] * gr[j * EEXP + e];
#pragma unroll
    for (int e = 0; e < EEXP; e++) red[tid * EEXP + e] = acc[e];
    __syncthreads();
    for (int s = 128; s > 0; s >>= 1) {
        if (tid < s)
#pragma unroll
            for (int e = 0; e < EEXP; e++)
                red[tid * EEXP + e] += red[(tid + s) * EEXP + e];
        __syncthreads();
    }
    if (tid == 0) {
        float lg[EEXP];
#pragma unroll
        for (int e = 0; e < EEXP; e++) {
            lg[e] = (1.0f - alpha) * (red[e] + gb[e]) + alpha * tlog[b * EEXP + e];
            glog[(size_t)t * EEXP + e] = lg[e];
        }
        int e0 = 0;
#pragma unroll
        for (int e = 1; e < EEXP; e++) if (lg[e] > lg[e0]) e0 = e;
        int e1 = (e0 == 0) ? 1 : 0;
#pragma unroll
        for (int e = 0; e < EEXP; e++)
            if (e != e0 && lg[e] > lg[e1]) e1 = e;
        float d = expf(lg[e1] - lg[e0]);       // <= 1
        float w0 = 1.0f / (1.0f + d);
        float w1 = d / (1.0f + d);
        sel[2 * t] = e0; sel[2 * t + 1] = e1;
        wgt[2 * t] = w0; wgt[2 * t + 1] = w1;
        atomicAdd(&counts[e0], 1);
        atomicAdd(&counts[e1], 1);
    }
}

__global__ void k_offsets(const int* __restrict__ counts, int* __restrict__ offs,
                          int* __restrict__ curs) {
    if (threadIdx.x == 0 && blockIdx.x == 0) {
        int off = 0;
        for (int e = 0; e < EEXP; e++) {
            offs[e] = off; curs[e] = off;
            off += (counts[e] + PADR - 1) & ~(PADR - 1);  // pad bucket to 32 rows
        }
        offs[EEXP] = off;
    }
}

__global__ void k_fill_list(int* __restrict__ list) {
    int i = blockIdx.x * blockDim.x + threadIdx.x;
    if (i < SLOT_MAX) list[i] = -1;
}

__global__ void k_scatter(const int* __restrict__ sel, int* __restrict__ curs,
                          int* __restrict__ list) {
    int t = blockIdx.x * blockDim.x + threadIdx.x;
    if (t < NTOK) {
#pragma unroll
        for (int k = 0; k < KTOP; k++) {
            int e = sel[2 * t + k];
            int pos = atomicAdd(&curs[e], 1);
            list[pos] = 2 * t + k;
        }
    }
}

// ---------------- WMMA grouped GEMMs ----------------
// Block = 128 threads (4 waves). Wave tile = 32 rows x 64 cols:
//   2 A fragments x 4 B fragments -> 8 WMMAs per K-step; A via 4 b128 loads, B via 64 strided 16-bit loads.
// Block tile = 32 rows x 256 cols.

__global__ __launch_bounds__(128)
void k_gemm1(const unsigned short* __restrict__ xb,
             const unsigned short* __restrict__ w1b,
             const int* __restrict__ list, const int* __restrict__ offs,
             unsigned short* __restrict__ hbuf) {
    const int slot0 = blockIdx.y * PADR;
    if (slot0 >= offs[EEXP]) return;          // uniform exit
    int e = 0;
#pragma unroll
    for (int i = 1; i < EEXP; i++) if (slot0 >= offs[i]) e = i;

    const int lane = threadIdx.x & 31;
    const int wave = threadIdx.x >> 5;
    const int n0 = blockIdx.x * 256 + wave * 64;
    const int m = lane & 15;
    const int ko = (lane >> 4) * 8;

    int ent0 = list[slot0 + m];
    int ent1 = list[slot0 + 16 + m];
    int tok0 = (ent0 >= 0) ? (ent0 >> 1) : 0;     // select, no EXEC divergence
    int tok1 = (ent1 >= 0) ? (ent1 >> 1) : 0;
    const unsigned short* arow0 = xb + (size_t)tok0 * DDIM + ko;
    const unsigned short* arow1 = xb + (size_t)tok1 * DDIM + ko;
    const unsigned short* brow  = w1b + (size_t)e * DDIM * HDIM + (size_t)ko * HDIM + n0 + m;

    v8f c[2][4] = {};
    for (int k = 0; k < DDIM; k += 32) {
        v16bf a0 = load_afrag(arow0 + k);
        v16bf a1 = load_afrag(arow1 + k);
        v16bf bf[4];
#pragma unroll
        for (int cg = 0; cg < 4; cg++)
        {
#pragma unroll
            for (int s = 0; s < 16; s++)
                bf[cg][s] = __builtin_bit_cast(__bf16, brow[(size_t)(k + (s < 8 ? s : s + 8)) * HDIM + 16 * cg]);
        }
#pragma unroll
        for (int cg = 0; cg < 4; cg++) {
            c[0][cg] = WMMA_BF16(a0, bf[cg], c[0][cg]);
            c[1][cg] = WMMA_BF16(a1, bf[cg], c[1][cg]);
        }
    }
    // C layout: lane -> N = lane&15; VGPR j -> M = j + (lane>>4)*8
    const int n = lane & 15;
    const int mbase = (lane >> 4) * 8;
#pragma unroll
    for (int j = 0; j < 8; j++) {
        int s0 = slot0 + mbase + j;
        int s1 = s0 + 16;
        if (list[s0] >= 0) {
#pragma unroll
            for (int cg = 0; cg < 4; cg++)
                hbuf[(size_t)s0 * HDIM + n0 + 16 * cg + n] = f2bf(gelu_tanh(c[0][cg][j]));
        }
        if (list[s1] >= 0) {
#pragma unroll
            for (int cg = 0; cg < 4; cg++)
                hbuf[(size_t)s1 * HDIM + n0 + 16 * cg + n] = f2bf(gelu_tanh(c[1][cg][j]));
        }
    }
}

__global__ __launch_bounds__(128)
void k_gemm2(const unsigned short* __restrict__ hbuf,
             const unsigned short* __restrict__ w2b,
             const int* __restrict__ list, const int* __restrict__ offs,
             const float* __restrict__ wgt, float* __restrict__ part) {
    const int slot0 = blockIdx.y * PADR;
    if (slot0 >= offs[EEXP]) return;
    int e = 0;
#pragma unroll
    for (int i = 1; i < EEXP; i++) if (slot0 >= offs[i]) e = i;

    const int lane = threadIdx.x & 31;
    const int wave = threadIdx.x >> 5;
    const int n0 = blockIdx.x * 256 + wave * 64;
    const int m = lane & 15;
    const int ko = (lane >> 4) * 8;

    const unsigned short* arow0 = hbuf + (size_t)(slot0 + m) * HDIM + ko;
    const unsigned short* arow1 = hbuf + (size_t)(slot0 + 16 + m) * HDIM + ko;
    const unsigned short* brow  = w2b + (size_t)e * HDIM * DDIM + (size_t)ko * DDIM + n0 + m;

    v8f c[2][4] = {};
    for (int k = 0; k < HDIM; k += 32) {
        v16bf a0 = load_afrag(arow0 + k);
        v16bf a1 = load_afrag(arow1 + k);
        v16bf bf[4];
#pragma unroll
        for (int cg = 0; cg < 4; cg++)
        {
#pragma unroll
            for (int s = 0; s < 16; s++)
                bf[cg][s] = __builtin_bit_cast(__bf16, brow[(size_t)(k + (s < 8 ? s : s + 8)) * DDIM + 16 * cg]);
        }
#pragma unroll
        for (int cg = 0; cg < 4; cg++) {
            c[0][cg] = WMMA_BF16(a0, bf[cg], c[0][cg]);
            c[1][cg] = WMMA_BF16(a1, bf[cg], c[1][cg]);
        }
    }
    const int n = lane & 15;
    const int mbase = (lane >> 4) * 8;
#pragma unroll
    for (int j = 0; j < 8; j++) {
        int s0 = slot0 + mbase + j;
        int s1 = s0 + 16;
        int en0 = list[s0];
        int en1 = list[s1];
        if (en0 >= 0) {
            float w = wgt[en0];
#pragma unroll
            for (int cg = 0; cg < 4; cg++)
                part[(size_t)en0 * DDIM + n0 + 16 * cg + n] = w * c[0][cg][j];
        }
        if (en1 >= 0) {
            float w = wgt[en1];
#pragma unroll
            for (int cg = 0; cg < 4; cg++)
                part[(size_t)en1 * DDIM + n0 + 16 * cg + n] = w * c[1][cg][j];
        }
    }
}

__global__ void k_combine(const float* __restrict__ part, float* __restrict__ out) {
    int i = blockIdx.x * blockDim.x + threadIdx.x;   // i = t*D + d
    int t = i >> 10;
    int d = i & (DDIM - 1);
    out[i] = part[((size_t)(2 * t)) * DDIM + d] + part[((size_t)(2 * t + 1)) * DDIM + d];
}

__global__ __launch_bounds__(256)
void k_laux(const float* __restrict__ glog, const int* __restrict__ sel,
            float* __restrict__ out) {
    __shared__ float sw[256 * EEXP];
    __shared__ float sc[256 * EEXP];
    const int tid = threadIdx.x;
    float aw[EEXP] = {}, ac[EEXP] = {};
    for (int t = tid; t < NTOK; t += 256) {
        float lg[EEXP], mx = -1e30f;
#pragma unroll
        for (int e = 0; e < EEXP; e++) { lg[e] = glog[(size_t)t * EEXP + e]; mx = fmaxf(mx, lg[e]); }
        float s = 0.0f;
#pragma unroll
        for (int e = 0; e < EEXP; e++) { lg[e] = expf(lg[e] - mx); s += lg[e]; }
        float inv = 1.0f / s;
#pragma unroll
        for (int e = 0; e < EEXP; e++) aw[e] += lg[e] * inv;
        ac[sel[2 * t]] += 1.0f;
        ac[sel[2 * t + 1]] += 1.0f;
    }
#pragma unroll
    for (int e = 0; e < EEXP; e++) { sw[tid * EEXP + e] = aw[e]; sc[tid * EEXP + e] = ac[e]; }
    __syncthreads();
    for (int s = 128; s > 0; s >>= 1) {
        if (tid < s)
#pragma unroll
            for (int e = 0; e < EEXP; e++) {
                sw[tid * EEXP + e] += sw[(tid + s) * EEXP + e];
                sc[tid * EEXP + e] += sc[(tid + s) * EEXP + e];
            }
        __syncthreads();
    }
    if (tid == 0) {
        float l = 0.0f;
#pragma unroll
        for (int e = 0; e < EEXP; e++) l += sw[e] * sc[e];
        out[0] = l * (1.0f / ((float)NTOK * (float)NTOK));
    }
}

// ---------------- launch ----------------

extern "C" void kernel_launch(void* const* d_in, const int* in_sizes, int n_in,
                              void* d_out, int out_size, void* d_ws, size_t ws_size,
                              hipStream_t stream) {
    (void)in_sizes; (void)n_in; (void)out_size; (void)ws_size;
    const float* x       = (const float*)d_in[0];
    const float* tp      = (const float*)d_in[1];
    const float* gw_img  = (const float*)d_in[2];
    const float* gb_img  = (const float*)d_in[3];
    const float* gw_txt  = (const float*)d_in[4];
    const float* gb_txt  = (const float*)d_in[5];
    const float* tg_w    = (const float*)d_in[6];
    const float* tg_b    = (const float*)d_in[7];
    const float* alpha   = (const float*)d_in[8];
    const float* w1      = (const float*)d_in[9];
    const float* w2      = (const float*)d_in[10];
    const int*   is_text = (const int*)d_in[11];
    float* out = (float*)d_out;

    // workspace layout (256-aligned chunks)
    constexpr size_t SZ_XB   = (size_t)NTOK * DDIM * 2;          // 8 MB
    constexpr size_t SZ_W    = (size_t)EEXP * DDIM * HDIM * 2;   // 64 MB each
    constexpr size_t SZ_H    = (size_t)SLOT_MAX * HDIM * 2;      // 66 MB
    constexpr size_t SZ_PART = (size_t)NENT * DDIM * 4;          // 32 MB
    constexpr size_t OFF_XB   = 0;
    constexpr size_t OFF_W1B  = OFF_XB + SZ_XB;
    constexpr size_t OFF_W2B  = OFF_W1B + SZ_W;
    constexpr size_t OFF_H    = OFF_W2B + SZ_W;
    constexpr size_t OFF_PART = OFF_H + SZ_H;
    constexpr size_t OFF_GLOG = OFF_PART + SZ_PART;
    constexpr size_t OFF_SEL  = OFF_GLOG + (size_t)NTOK * EEXP * 4;
    constexpr size_t OFF_WGT  = OFF_SEL + (size_t)NENT * 4;
    constexpr size_t OFF_TLOG = OFF_WGT + (size_t)NENT * 4;
    constexpr size_t OFF_CNT  = OFF_TLOG + 256;
    constexpr size_t OFF_CUR  = OFF_CNT + 256;
    constexpr size_t OFF_OFFS = OFF_CUR + 256;
    constexpr size_t OFF_LIST = OFF_OFFS + 256;

    char* ws = (char*)d_ws;
    unsigned short* xb   = (unsigned short*)(ws + OFF_XB);
    unsigned short* w1b  = (unsigned short*)(ws + OFF_W1B);
    unsigned short* w2b  = (unsigned short*)(ws + OFF_W2B);
    unsigned short* hbuf = (unsigned short*)(ws + OFF_H);
    float* part  = (float*)(ws + OFF_PART);
    float* glog  = (float*)(ws + OFF_GLOG);
    int*   sel   = (int*)(ws + OFF_SEL);
    float* wgt   = (float*)(ws + OFF_WGT);
    float* tlog  = (float*)(ws + OFF_TLOG);
    int*   cnt   = (int*)(ws + OFF_CNT);
    int*   cur   = (int*)(ws + OFF_CUR);
    int*   offs  = (int*)(ws + OFF_OFFS);
    int*   list  = (int*)(ws + OFF_LIST);

    k_zero_counts<<<1, 32, 0, stream>>>(cnt);
    k_cvt_bf16<<<4096, 256, 0, stream>>>(x, xb, NTOK * DDIM / 4);
    k_cvt_bf16<<<8192, 256, 0, stream>>>(w1, w1b, EEXP * DDIM * HDIM / 4);
    k_cvt_bf16<<<8192, 256, 0, stream>>>(w2, w2b, EEXP * HDIM * DDIM / 4);
    k_task_logits<<<BDIM, 256, 0, stream>>>(tp, tg_w, tg_b, tlog);
    k_gate<<<NTOK, 256, 0, stream>>>(x, gw_img, gb_img, gw_txt, gb_txt,
                                     alpha, is_text, tlog, glog, sel, wgt, cnt);
    k_offsets<<<1, 1, 0, stream>>>(cnt, offs, cur);
    k_fill_list<<<(SLOT_MAX + 255) / 256, 256, 0, stream>>>(list);
    k_scatter<<<NTOK / 256, 256, 0, stream>>>(sel, cur, list);
    k_gemm1<<<dim3(HDIM / 256, NTILES), 128, 0, stream>>>(xb, w1b, list, offs, hbuf);
    k_gemm2<<<dim3(DDIM / 256, NTILES), 128, 0, stream>>>(hbuf, w2b, list, offs, wgt, part);
    k_combine<<<(NTOK * DDIM) / 256, 256, 0, stream>>>(part, out);
    k_laux<<<1, 256, 0, stream>>>(glog, sel, out + (size_t)NTOK * DDIM);
}
